// HoloBlock_68272800137366
// MI455X (gfx1250) — matrix operand order repaired and checked by reading, not compiled, not run on hardware
//
#include <hip/hip_runtime.h>
#include <math.h>

// ---------------- problem constants ----------------
#define B_   2
#define T_   4096
#define C_   1024
#define H_   16
#define HD_  64
#define FF_  4096
#define KS_  4
#define M_   (B_*T_)          // 8192 rows
#define NCH  32               // scan chunks
#define TCH  (T_/NCH)         // 128

typedef float  v8f     __attribute__((ext_vector_type(8)));
typedef __bf16 bf16x16 __attribute__((ext_vector_type(16)));
typedef __bf16 bf16x8  __attribute__((ext_vector_type(8)));

// ---------------- f32 -> bf16 weight convert ----------------
__global__ void cvt_f32_bf16(const float* __restrict__ s, __bf16* __restrict__ d, int n) {
    int i = blockIdx.x * blockDim.x + threadIdx.x;
    int st = gridDim.x * blockDim.x;
    for (; i < n; i += st) d[i] = (__bf16)s[i];
}

// ---------------- LN1: h = LN(x) ----------------
__global__ __launch_bounds__(256)
void ln1_kernel(const float* __restrict__ x, const float* __restrict__ w,
                const float* __restrict__ b, float* __restrict__ out) {
    __shared__ float sm[256];
    const int r = blockIdx.x, t = threadIdx.x;
    const size_t base = (size_t)r * C_;
    float a[4]; float s = 0.f;
#pragma unroll
    for (int i = 0; i < 4; ++i) { a[i] = x[base + t + 256*i]; s += a[i]; }
    sm[t] = s; __syncthreads();
    for (int k = 128; k > 0; k >>= 1) { if (t < k) sm[t] += sm[t+k]; __syncthreads(); }
    float mean = sm[0] * (1.f/C_); __syncthreads();
    float vs = 0.f;
#pragma unroll
    for (int i = 0; i < 4; ++i) { float d = a[i]-mean; vs += d*d; }
    sm[t] = vs; __syncthreads();
    for (int k = 128; k > 0; k >>= 1) { if (t < k) sm[t] += sm[t+k]; __syncthreads(); }
    float rstd = rsqrtf(sm[0] * (1.f/C_) + 1e-5f);
#pragma unroll
    for (int i = 0; i < 4; ++i) {
        int c = t + 256*i;
        out[base + c] = (a[i]-mean)*rstd*w[c] + b[c];
    }
}

// ---------------- depthwise causal conv -> bf16 A matrix ----------------
__global__ void conv_kernel(const float* __restrict__ h, const float* __restrict__ cw,
                            const float* __restrict__ cb, __bf16* __restrict__ xc) {
    size_t i = (size_t)blockIdx.x * blockDim.x + threadIdx.x;
    if (i >= (size_t)M_ * C_) return;
    int c = (int)(i % C_);
    int t = (int)((i / C_) % T_);
    float acc = cb[c];
#pragma unroll
    for (int j = 0; j < KS_; ++j) {
        int ts = t - (KS_-1) + j;
        if (ts >= 0) acc += h[i + (size_t)(j - (KS_-1)) * C_] * cw[c*KS_ + j];
    }
    xc[i] = (__bf16)acc;
}

// ---------------- tiled bf16 WMMA GEMM: C[M,N] = A[M,K] @ W[N,K]^T (+bias) ----------------
// Double-buffered LDS pipeline: one barrier per K-step; next tile's global loads
// are issued before the current tile's 8 WMMAs so HBM/L2 latency overlaps compute.
#define TBM 128
#define TBN 128
#define TBK 32
#define LDP (TBK + 8)   // padded LDS row (bf16 elems) to stagger banks

__global__ __launch_bounds__(256)
void gemm_bf16_kernel(const __bf16* __restrict__ A, const __bf16* __restrict__ W,
                      const float* __restrict__ bias, void* __restrict__ Cout,
                      int M, int N, int K, int out_bf16) {
    __shared__ __bf16 As[2][TBM * LDP];
    __shared__ __bf16 Bs[2][TBN * LDP];

    const int tid  = threadIdx.x;
    const int lane = tid & 31;
    const int wid  = tid >> 5;     // 0..7
    const int wr   = wid >> 1;     // 0..3 : 32-row group
    const int wc   = wid & 1;      // 0..1 : 64-col group
    const int m0   = blockIdx.y * TBM;
    const int n0   = blockIdx.x * TBN;

    const v8f vzero = {0.f,0.f,0.f,0.f,0.f,0.f,0.f,0.f};
    v8f acc[2][4];
#pragma unroll
    for (int i = 0; i < 2; ++i)
#pragma unroll
        for (int j = 0; j < 4; ++j) acc[i][j] = vzero;

    // tile staging: 256 threads, each moves 16 bf16 (two 16B chunks) of A and of B
    const int lrow = tid >> 1;
    const int lcol = (tid & 1) * 16;
    const __bf16* ga = A + (size_t)(m0 + lrow) * K + lcol;
    const __bf16* gb = W + (size_t)(n0 + lrow) * K + lcol;
    const int soff = lrow * LDP + lcol;

    // prologue: stage tile 0 into buffer 0
    {
        bf16x8 a0 = *(const bf16x8*)ga;
        bf16x8 a1 = *(const bf16x8*)(ga + 8);
        bf16x8 b0 = *(const bf16x8*)gb;
        bf16x8 b1 = *(const bf16x8*)(gb + 8);
        *(bf16x8*)&As[0][soff]     = a0;
        *(bf16x8*)&As[0][soff + 8] = a1;
        *(bf16x8*)&Bs[0][soff]     = b0;
        *(bf16x8*)&Bs[0][soff + 8] = b1;
    }

    const int nk = K / TBK;
    int cur = 0;
    for (int kt = 0; kt < nk; ++kt) {
        __syncthreads();   // buffer[cur] visible to all waves

        // issue next tile's global loads early (latency hidden behind WMMAs below)
        bf16x8 a0 = {}, a1 = {}, b0 = {}, b1 = {};
        const bool more = (kt + 1 < nk);
        if (more) {
            const __bf16* gan = ga + (size_t)(kt + 1) * TBK;
            const __bf16* gbn = gb + (size_t)(kt + 1) * TBK;
            a0 = *(const bf16x8*)gan; a1 = *(const bf16x8*)(gan + 8);
            b0 = *(const bf16x8*)gbn; b1 = *(const bf16x8*)(gbn + 8);
            if (kt + 2 < nk) {                   // -> global_prefetch_b8 (tile k+2)
                __builtin_prefetch(gan + TBK, 0, 1);
                __builtin_prefetch(gbn + TBK, 0, 1);
            }
        }

        // A fragment: lane<16 holds row(lane), K {kb..kb+7, kb+16..kb+23}, kb = (lane>>4)*8
        const int arow = wr * 32 + (lane & 15);
        const int kbA  = (lane >> 4) * 8;
        bf16x16 afrag[2];
#pragma unroll
        for (int i = 0; i < 2; ++i) {
            const __bf16* p = &As[cur][(arow + i*16) * LDP + kbA];
            bf16x8 lo = *(const bf16x8*)p;
            bf16x8 hi = *(const bf16x8*)(p + 16);
            afrag[i] = __builtin_shufflevector(lo, hi, 0,1,2,3,4,5,6,7,8,9,10,11,12,13,14,15);
        }
        // B fragment: lane<16 -> col(lane), K {kb..kb+7, kb+16..kb+23}, kb = (lane>>4)*8
        const int bcol = wc * 64 + (lane & 15);
        const int kbB  = (lane >> 4) * 8;
        bf16x16 bfrag[4];
#pragma unroll
        for (int j = 0; j < 4; ++j) {
            const __bf16* p = &Bs[cur][(bcol + j*16) * LDP + kbB];
            bf16x8 lo = *(const bf16x8*)p;
            bf16x8 hi = *(const bf16x8*)(p + 16);
            bfrag[j] = __builtin_shufflevector(lo, hi, 0,1,2,3,4,5,6,7,8,9,10,11,12,13,14,15);
        }
#pragma unroll
        for (int i = 0; i < 2; ++i)
#pragma unroll
            for (int j = 0; j < 4; ++j)
                acc[i][j] = __builtin_amdgcn_wmma_f32_16x16x32_bf16(
                    false, afrag[i], false, bfrag[j], (short)0, acc[i][j], false, false);

        // stage next tile into the other buffer (no reader conflicts: all reads of
        // buffer[cur^1] completed before the barrier at the top of this iteration)
        if (more) {
            const int nxt = cur ^ 1;
            *(bf16x8*)&As[nxt][soff]     = a0;
            *(bf16x8*)&As[nxt][soff + 8] = a1;
            *(bf16x8*)&Bs[nxt][soff]     = b0;
            *(bf16x8*)&Bs[nxt][soff + 8] = b1;
        }
        cur ^= 1;
    }

    // epilogue: element e of v8f -> row (lane>>4)*8 + e, col lane&15
    const int colb = n0 + wc * 64 + (lane & 15);
    const int rowb = m0 + wr * 32 + ((lane >> 4) * 8);
#pragma unroll
    for (int i = 0; i < 2; ++i)
#pragma unroll
        for (int j = 0; j < 4; ++j) {
            int c = colb + j * 16;
            float bv = bias ? bias[c] : 0.f;
#pragma unroll
            for (int e = 0; e < 8; ++e) {
                int r = rowb + i * 16 + e;
                float val = acc[i][j][e] + bv;
                if (out_bf16) ((__bf16*)Cout)[(size_t)r * N + c] = (__bf16)val;
                else          ((float*)Cout)[(size_t)r * N + c]  = val;
            }
        }
}

// ---------------- k = qk / max(||qk||,1e-12) per (b,t,h) ----------------
__global__ __launch_bounds__(64)
void knorm_kernel(float* __restrict__ qk) {
    __shared__ float sm[64];
    const size_t base = (size_t)blockIdx.x * HD_;
    const int d = threadIdx.x;
    float v = qk[base + d];
    sm[d] = v * v; __syncthreads();
    for (int k = 32; k > 0; k >>= 1) { if (d < k) sm[d] += sm[d+k]; __syncthreads(); }
    qk[base + d] = v / fmaxf(sqrtf(sm[0]), 1e-12f);
}

// ---------------- g = sigmoid(g_lin) in place ----------------
__global__ void sigmoid_kernel(float* __restrict__ g, size_t n) {
    size_t i = (size_t)blockIdx.x * blockDim.x + threadIdx.x;
    if (i < n) g[i] = 1.f / (1.f + expf(-g[i]));
}

// ---------------- scan phase 1: build w[t]=g*v*(cos(t*f)+k[t-1])*gamma^t, chunk sums ----------------
__global__ __launch_bounds__(64)
void scan_build_kernel(const float* __restrict__ v, const float* __restrict__ k,
                       const float* __restrict__ g, const float* __restrict__ gam,
                       float* __restrict__ w, float* __restrict__ csum) {
    const int blk = blockIdx.x;
    const int ch = blk % NCH;
    const int hh = (blk / NCH) % H_;
    const int b  = blk / (NCH * H_);
    const int d  = threadIdx.x;
    const float gamma = gam[hh];
    const float freq  = powf(10.f, (float)d * (1.f/(HD_-1))) * (float)hh;
    const int   t0    = ch * TCH;
    float decay = powf(gamma, (float)t0);
    const size_t stride = (size_t)H_ * HD_;
    const size_t base   = ((size_t)b * T_ + t0) * stride + (size_t)hh * HD_ + d;
    float s = 0.f;
    for (int i = 0; i < TCH; ++i) {
        const int t = t0 + i;
        const size_t ix = base + (size_t)i * stride;
        float ks = (t == 0) ? 0.f : k[ix - stride];
        float wv = g[ix] * v[ix] * (cosf((float)t * freq) + ks) * decay;
        w[ix] = wv; s += wv;
        decay *= gamma;
    }
    csum[((size_t)((b*H_ + hh) * HD_ + d)) * NCH + ch] = s;
}

// ---------------- scan phase 2: exclusive scan of chunk sums per series ----------------
__global__ void scan_chunks_kernel(float* __restrict__ csum) {
    int s = blockIdx.x * blockDim.x + threadIdx.x;
    if (s >= B_ * H_ * HD_) return;
    size_t base = (size_t)s * NCH;
    float run = 0.f;
    for (int c = 0; c < NCH; ++c) { float v = csum[base + c]; csum[base + c] = run; run += v; }
}

// ---------------- scan phase 3: out = (cumsum/ (gamma^t+1e-6)) * q  -> bf16 A for o-GEMM --------
__global__ __launch_bounds__(64)
void scan_apply_kernel(const float* __restrict__ w, const float* __restrict__ k,
                       const float* __restrict__ csum, const float* __restrict__ gam,
                       __bf16* __restrict__ outb) {
    const int blk = blockIdx.x;
    const int ch = blk % NCH;
    const int hh = (blk / NCH) % H_;
    const int b  = blk / (NCH * H_);
    const int d  = threadIdx.x;
    const float gamma = gam[hh];
    const int   t0    = ch * TCH;
    float decay = powf(gamma, (float)t0);
    float run = csum[((size_t)((b*H_ + hh) * HD_ + d)) * NCH + ch];
    const size_t stride = (size_t)H_ * HD_;
    const size_t base   = ((size_t)b * T_ + t0) * stride + (size_t)hh * HD_ + d;
    for (int i = 0; i < TCH; ++i) {
        const size_t ix = base + (size_t)i * stride;
        run += w[ix];
        float mem = run / (decay + 1e-6f);
        outb[ix] = (__bf16)(mem * k[ix]);
        decay *= gamma;
    }
}

// ---------------- post-attn: LN(o) epilogue + residual + LN2 -> bf16 MLP input ----------------
__global__ __launch_bounds__(256)
void post_attn_kernel(const float* __restrict__ alin, const float* __restrict__ x,
                      const float* __restrict__ g1,
                      const float* __restrict__ ow, const float* __restrict__ ob,
                      const float* __restrict__ l2w, const float* __restrict__ l2b,
                      float* __restrict__ x2, __bf16* __restrict__ h2) {
    __shared__ float sm[256];
    const int r = blockIdx.x, t = threadIdx.x;
    const size_t base = (size_t)r * C_;
    float a[4]; float s = 0.f;
#pragma unroll
    for (int i = 0; i < 4; ++i) { a[i] = alin[base + t + 256*i]; s += a[i]; }
    sm[t] = s; __syncthreads();
    for (int kk = 128; kk > 0; kk >>= 1) { if (t < kk) sm[t] += sm[t+kk]; __syncthreads(); }
    float mean = sm[0] * (1.f/C_); __syncthreads();
    float vs = 0.f;
#pragma unroll
    for (int i = 0; i < 4; ++i) { float d = a[i]-mean; vs += d*d; }
    sm[t] = vs; __syncthreads();
    for (int kk = 128; kk > 0; kk >>= 1) { if (t < kk) sm[t] += sm[t+kk]; __syncthreads(); }
    float rstd = rsqrtf(sm[0] * (1.f/C_) + 1e-5f); __syncthreads();

    float xv[4]; s = 0.f;
#pragma unroll
    for (int i = 0; i < 4; ++i) {
        int c = t + 256*i;
        float an = (a[i]-mean)*rstd*ow[c] + ob[c];
        xv[i] = x[base + c] + g1[c]*an;
        x2[base + c] = xv[i];
        s += xv[i];
    }
    sm[t] = s; __syncthreads();
    for (int kk = 128; kk > 0; kk >>= 1) { if (t < kk) sm[t] += sm[t+kk]; __syncthreads(); }
    float mean2 = sm[0] * (1.f/C_); __syncthreads();
    vs = 0.f;
#pragma unroll
    for (int i = 0; i < 4; ++i) { float d = xv[i]-mean2; vs += d*d; }
    sm[t] = vs; __syncthreads();
    for (int kk = 128; kk > 0; kk >>= 1) { if (t < kk) sm[t] += sm[t+kk]; __syncthreads(); }
    float rstd2 = rsqrtf(sm[0] * (1.f/C_) + 1e-5f);
#pragma unroll
    for (int i = 0; i < 4; ++i) {
        int c = t + 256*i;
        h2[base + c] = (__bf16)((xv[i]-mean2)*rstd2*l2w[c] + l2b[c]);
    }
}

// ---------------- act = silu(gate)*up (bf16, in place over gate) ----------------
__global__ void swiglu_kernel(__bf16* __restrict__ gate, const __bf16* __restrict__ up, size_t n) {
    size_t i = (size_t)blockIdx.x * blockDim.x + threadIdx.x;
    if (i >= n) return;
    float a = (float)gate[i];
    float u = (float)up[i];
    float sv = a / (1.f + expf(-a));
    gate[i] = (__bf16)(sv * u);
}

// ---------------- final residual ----------------
__global__ void final_kernel(const float* __restrict__ x2, const float* __restrict__ mlp,
                             const float* __restrict__ g2, float* __restrict__ out, size_t n) {
    size_t i = (size_t)blockIdx.x * blockDim.x + threadIdx.x;
    if (i >= n) return;
    int c = (int)(i % C_);
    out[i] = x2[i] + g2[c] * mlp[i];
}

// =====================================================================
extern "C" void kernel_launch(void* const* d_in, const int* in_sizes, int n_in,
                              void* d_out, int out_size, void* d_ws, size_t ws_size,
                              hipStream_t stream) {
    (void)in_sizes; (void)n_in; (void)out_size; (void)ws_size;
    const float* x        = (const float*)d_in[0];
    const float* ln1_w    = (const float*)d_in[1];
    const float* ln1_b    = (const float*)d_in[2];
    const float* conv_w   = (const float*)d_in[3];
    const float* conv_b   = (const float*)d_in[4];
    const float* qk_w     = (const float*)d_in[5];
    const float* v_w      = (const float*)d_in[6];
    const float* g_w      = (const float*)d_in[7];
    const float* g_b      = (const float*)d_in[8];
    const float* o_w      = (const float*)d_in[9];
    const float* onorm_w  = (const float*)d_in[10];
    const float* onorm_b  = (const float*)d_in[11];
    const float* gamma_at = (const float*)d_in[12];
    const float* ln2_w    = (const float*)d_in[13];
    const float* ln2_b    = (const float*)d_in[14];
    const float* gate_w   = (const float*)d_in[15];
    const float* up_w     = (const float*)d_in[16];
    const float* down_w   = (const float*)d_in[17];
    const float* gamma1   = (const float*)d_in[18];
    const float* gamma2   = (const float*)d_in[19];
    float* out = (float*)d_out;

    const size_t nW1  = (size_t)C_ * C_;        // 1M  (qk,v,g,o)
    const size_t nWff = (size_t)FF_ * C_;       // 4M  (gate,up,down)
    const size_t nAct = (size_t)M_ * C_;        // 8.39M
    const size_t nFF  = (size_t)M_ * FF_;       // 33.5M

    // ---- workspace carve-out (manual reuse; ~320 MB) ----
    char* wsb = (char*)d_ws;
    size_t cur = 0;
    auto alloc = [&](size_t bytes) -> void* {
        void* p = wsb + cur;
        cur += (bytes + 255) & ~(size_t)255;
        return p;
    };
    __bf16* qk_bf   = (__bf16*)alloc(nW1 * 2);
    __bf16* v_bf    = (__bf16*)alloc(nW1 * 2);
    __bf16* g_bf    = (__bf16*)alloc(nW1 * 2);
    __bf16* o_bf    = (__bf16*)alloc(nW1 * 2);
    __bf16* gate_wb = (__bf16*)alloc(nWff * 2);
    __bf16* up_wb   = (__bf16*)alloc(nWff * 2);
    __bf16* down_wb = (__bf16*)alloc(nWff * 2);
    float*  h_f32   = (float*)alloc(nAct * 4);   // h; reused: w (scan), mlp out
    __bf16* xc_bf   = (__bf16*)alloc(nAct * 2);  // conv out; reused: attn out (o-GEMM A)
    float*  k_f32   = (float*)alloc(nAct * 4);   // qk -> k; reused: h2 bf16
    float*  v_f32   = (float*)alloc(nAct * 4);   // v; reused: attn_lin
    float*  g_f32   = (float*)alloc(nAct * 4);   // g; reused: x2
    float*  csum    = (float*)alloc((size_t)B_*H_*HD_*NCH * 4);
    __bf16* gateb   = (__bf16*)alloc(nFF * 2);   // gate lin; swiglu writes act in place
    __bf16* upb     = (__bf16*)alloc(nFF * 2);
    // aliases (lifetimes are disjoint — see launch order)
    float*  wbuf     = h_f32;
    float*  mlp_f32  = h_f32;
    __bf16* out_bf   = xc_bf;
    float*  attn_lin = v_f32;
    float*  x2       = g_f32;
    __bf16* h2_bf    = (__bf16*)k_f32;

    // 1) weights -> bf16
    cvt_f32_bf16<<<4096, 256, 0, stream>>>(qk_w,   qk_bf,   (int)nW1);
    cvt_f32_bf16<<<4096, 256, 0, stream>>>(v_w,    v_bf,    (int)nW1);
    cvt_f32_bf16<<<4096, 256, 0, stream>>>(g_w,    g_bf,    (int)nW1);
    cvt_f32_bf16<<<4096, 256, 0, stream>>>(o_w,    o_bf,    (int)nW1);
    cvt_f32_bf16<<<8192, 256, 0, stream>>>(gate_w, gate_wb, (int)nWff);
    cvt_f32_bf16<<<8192, 256, 0, stream>>>(up_w,   up_wb,   (int)nWff);
    cvt_f32_bf16<<<8192, 256, 0, stream>>>(down_w, down_wb, (int)nWff);

    // 2) h = LN1(x)
    ln1_kernel<<<M_, 256, 0, stream>>>(x, ln1_w, ln1_b, h_f32);
    // 3) xc = depthwise causal conv(h) -> bf16
    conv_kernel<<<(unsigned)(nAct/256), 256, 0, stream>>>(h_f32, conv_w, conv_b, xc_bf);

    // 4-6) qk / v / g projections (bf16 WMMA, f32 out)
    dim3 g1k(C_/TBN, M_/TBM);
    gemm_bf16_kernel<<<g1k, 256, 0, stream>>>(xc_bf, qk_bf, nullptr, k_f32, M_, C_, C_, 0);
    gemm_bf16_kernel<<<g1k, 256, 0, stream>>>(xc_bf, v_bf,  nullptr, v_f32, M_, C_, C_, 0);
    gemm_bf16_kernel<<<g1k, 256, 0, stream>>>(xc_bf, g_bf,  g_b,     g_f32, M_, C_, C_, 0);

    // 7) k normalize per (b,t,h); 8) g sigmoid
    knorm_kernel<<<(unsigned)(M_*H_), 64, 0, stream>>>(k_f32);
    sigmoid_kernel<<<(unsigned)(nAct/256), 256, 0, stream>>>(g_f32, nAct);

    // 9-11) decay-weighted prefix scan (real part only — imaginary path is dead code)
    const unsigned scanBlks = (unsigned)(B_ * H_ * NCH);
    scan_build_kernel<<<scanBlks, 64, 0, stream>>>(v_f32, k_f32, g_f32, gamma_at, wbuf, csum);
    scan_chunks_kernel<<<(B_*H_*HD_ + 255)/256, 256, 0, stream>>>(csum);
    scan_apply_kernel<<<scanBlks, 64, 0, stream>>>(wbuf, k_f32, csum, gamma_at, out_bf);

    // 12) o-projection
    gemm_bf16_kernel<<<g1k, 256, 0, stream>>>(out_bf, o_bf, nullptr, attn_lin, M_, C_, C_, 0);
    // 13) LN(o) + residual + LN2
    post_attn_kernel<<<M_, 256, 0, stream>>>(attn_lin, x, gamma1, onorm_w, onorm_b,
                                             ln2_w, ln2_b, x2, h2_bf);

    // 14-17) SwiGLU MLP
    dim3 gff(FF_/TBN, M_/TBM);
    gemm_bf16_kernel<<<gff, 256, 0, stream>>>(h2_bf, gate_wb, nullptr, gateb, M_, FF_, C_, 1);
    gemm_bf16_kernel<<<gff, 256, 0, stream>>>(h2_bf, up_wb,   nullptr, upb,   M_, FF_, C_, 1);
    swiglu_kernel<<<(unsigned)(nFF/256), 256, 0, stream>>>(gateb, upb, nFF);
    gemm_bf16_kernel<<<g1k, 256, 0, stream>>>(gateb, down_wb, nullptr, mlp_f32, M_, C_, FF_, 0);

    // 18) final residual
    final_kernel<<<(unsigned)(nAct/256), 256, 0, stream>>>(x2, mlp_f32, gamma2, out, nAct);
}
